// DefSeq_27358941675873
// MI455X (gfx1250) — matrix operand order repaired and checked by reading, not compiled, not run on hardware
//
#include <hip/hip_runtime.h>

#define DEV __device__ __forceinline__

typedef __attribute__((ext_vector_type(2)))  float  v2f;
typedef __attribute__((ext_vector_type(8)))  float  v8f;
typedef __attribute__((ext_vector_type(8)))  __bf16 v8bf;
typedef __attribute__((ext_vector_type(16))) __bf16 v16bf;

constexpr int Vv = 32000, Ee = 512, Hh = 1024, Ss = 32, Bb = 32;

DEV float sigm(float x) { return 1.0f / (1.0f + __expf(-x)); }

DEV v8f wmma_f32(v2f a, v2f b, v8f c) {
  // V_WMMA_F32_16X16X4_F32 : D = A(16x4 f32) * B(4x16 f32) + C
  return __builtin_amdgcn_wmma_f32_16x16x4_f32(false, a, false, b, (short)0, c,
                                               false, false);
}
DEV v8f wmma_bf16(v16bf a, v16bf b, v8f c) {
  // V_WMMA_F32_16X16X32_BF16
  return __builtin_amdgcn_wmma_f32_16x16x32_bf16(false, a, false, b, (short)0, c,
                                                 false, false);
}
DEV v16bf cat16(v8bf lo, v8bf hi) {
  return __builtin_shufflevector(lo, hi, 0, 1, 2, 3, 4, 5, 6, 7, 8, 9, 10, 11,
                                 12, 13, 14, 15);
}
// byte offset of a __shared__ address relative to the wave's LDS base
DEV unsigned lds_offset(const void* p) {
  return (unsigned)(unsigned long long)(__attribute__((address_space(3))) const void*)p;
}

// ---------------------------------------------------------------- gather we
__global__ __launch_bounds__(256) void k_gather_we(
    const float* __restrict__ emb, const int* __restrict__ word,
    float* __restrict__ we) {
  int i = blockIdx.x * 256 + threadIdx.x;     // 64 blocks -> 16384 = B*E
  int m = i >> 9, e = i & 511;
  we[i] = emb[(size_t)word[m] * Ee + e];
}

// ------------------------------------------------- h0 = we @ Wl^T + bl ; c=h0
__global__ __launch_bounds__(64) void k_init(
    const float* __restrict__ we, const float* __restrict__ Wl,
    const float* __restrict__ bl, float* __restrict__ h, float* __restrict__ c) {
  const int n0 = blockIdx.x * 16;                 // 64 blocks over H
  const int wave = threadIdx.x >> 5, lane = threadIdx.x & 31;
  const int half = lane >> 4, l16 = lane & 15;
  const int m0 = wave * 16;                       // 2 waves over B=32
  const float* arow = we + (m0 + l16) * Ee;
  const float* brow = Wl + (size_t)(n0 + l16) * Ee;
  v8f acc = {};
  for (int k = 0; k < Ee; k += 4) {
    v2f a = *(const v2f*)(arow + k + half * 2);
    v2f b = *(const v2f*)(brow + k + half * 2);
    acc = wmma_f32(a, b, acc);
  }
  float bias = bl[n0 + l16];
  for (int j = 0; j < 8; ++j) {
    int row = m0 + j + 8 * half;
    float v = acc[j] + bias;
    h[row * Hh + n0 + l16] = v;
    c[row * Hh + n0 + l16] = v;
  }
}

// ------------------------------------------ LSTM cell: gates + nonlinearity
__global__ __launch_bounds__(128) void k_lstm(
    const float* __restrict__ emb, const int* __restrict__ seq, int s,
    const float* __restrict__ h_in, float* __restrict__ c_io,
    const float* __restrict__ Wih, const float* __restrict__ Whh,
    const float* __restrict__ bih, const float* __restrict__ bhh,
    float* __restrict__ hl) {
  __shared__ float gsm[4][32][16];
  const int n0 = blockIdx.x * 16;                 // 64 blocks over H
  const int wave = threadIdx.x >> 5;              // gate index i,f,g,o
  const int lane = threadIdx.x & 31;
  const int half = lane >> 4, l16 = lane & 15;
  const int gn = wave * Hh + n0 + l16;            // row in Wih/Whh
  const float* bi = Wih + (size_t)gn * Ee;
  const float* bh2 = Whh + (size_t)gn * Hh;
  const float* x0 = emb + (size_t)seq[s * Bb + l16] * Ee;
  const float* x1 = emb + (size_t)seq[s * Bb + 16 + l16] * Ee;
  const float* h0r = h_in + l16 * Hh;
  const float* h1r = h_in + (16 + l16) * Hh;
  v8f acc0 = {}, acc1 = {};
  for (int k = 0; k < Ee; k += 4) {
    v2f b = *(const v2f*)(bi + k + half * 2);
    acc0 = wmma_f32(*(const v2f*)(x0 + k + half * 2), b, acc0);
    acc1 = wmma_f32(*(const v2f*)(x1 + k + half * 2), b, acc1);
  }
  for (int k = 0; k < Hh; k += 4) {
    v2f b = *(const v2f*)(bh2 + k + half * 2);
    acc0 = wmma_f32(*(const v2f*)(h0r + k + half * 2), b, acc0);
    acc1 = wmma_f32(*(const v2f*)(h1r + k + half * 2), b, acc1);
  }
  float bias = bih[gn] + bhh[gn];
  for (int j = 0; j < 8; ++j) {
    gsm[wave][j + 8 * half][l16] = acc0[j] + bias;
    gsm[wave][16 + j + 8 * half][l16] = acc1[j] + bias;
  }
  __syncthreads();
  for (int t = threadIdx.x; t < 512; t += 128) {
    int row = t >> 4, col = t & 15;
    float ig = gsm[0][row][col], fg = gsm[1][row][col];
    float gg = gsm[2][row][col], og = gsm[3][row][col];
    int idx = row * Hh + n0 + col;
    float cn = sigm(fg) * c_io[idx] + sigm(ig) * tanhf(gg);
    c_io[idx] = cn;                               // column-owned: safe in place
    hl[idx] = sigm(og) * tanhf(cn);
  }
}

// -------------------------------- z = sig([we|hl]@Wz^T+bz), r = sig(...Wr...)
__global__ __launch_bounds__(64) void k_zr(
    const float* __restrict__ we, const float* __restrict__ hl,
    const float* __restrict__ Wz, const float* __restrict__ bz,
    const float* __restrict__ Wr, const float* __restrict__ br,
    float* __restrict__ z, float* __restrict__ r) {
  const bool is_r = blockIdx.x >= 64;             // 64 z-tiles + 32 r-tiles
  const int n0 = (is_r ? (int)blockIdx.x - 64 : (int)blockIdx.x) * 16;
  const int wave = threadIdx.x >> 5, lane = threadIdx.x & 31;
  const int half = lane >> 4, l16 = lane & 15;
  const int m0 = wave * 16;
  const int n = n0 + l16;
  const float* brow = (is_r ? Wr : Wz) + (size_t)n * (Ee + Hh);
  const float* wer = we + (m0 + l16) * Ee;
  const float* hlr = hl + (m0 + l16) * Hh;
  v8f acc = {};
  for (int k = 0; k < Ee; k += 4)
    acc = wmma_f32(*(const v2f*)(wer + k + half * 2),
                   *(const v2f*)(brow + k + half * 2), acc);
  for (int k = 0; k < Hh; k += 4)
    acc = wmma_f32(*(const v2f*)(hlr + k + half * 2),
                   *(const v2f*)(brow + Ee + k + half * 2), acc);
  float bias = is_r ? br[n] : bz[n];
  for (int j = 0; j < 8; ++j) {
    int row = m0 + j + 8 * half;
    float v = sigm(acc[j] + bias);
    if (is_r) r[row * Ee + n] = v; else z[row * Hh + n] = v;
  }
}

// ----------------- hh = tanh([r*we|hl]@Wh^T+bh); h = (1-z)hl + z hh -> h, hs
__global__ __launch_bounds__(64) void k_hh(
    const float* __restrict__ we, const float* __restrict__ hl,
    const float* __restrict__ rb, const float* __restrict__ zb,
    const float* __restrict__ Wh, const float* __restrict__ bh,
    float* __restrict__ h_out, float* __restrict__ hs_s) {
  const int n0 = blockIdx.x * 16;
  const int wave = threadIdx.x >> 5, lane = threadIdx.x & 31;
  const int half = lane >> 4, l16 = lane & 15;
  const int m0 = wave * 16;
  const int n = n0 + l16;
  const float* brow = Wh + (size_t)n * (Ee + Hh);
  const float* wer = we + (m0 + l16) * Ee;
  const float* rr = rb + (m0 + l16) * Ee;
  const float* hlr = hl + (m0 + l16) * Hh;
  v8f acc = {};
  for (int k = 0; k < Ee; k += 4) {
    int kk = k + half * 2;
    v2f a; a.x = rr[kk] * wer[kk]; a.y = rr[kk + 1] * wer[kk + 1];
    acc = wmma_f32(a, *(const v2f*)(brow + kk), acc);
  }
  for (int k = 0; k < Hh; k += 4)
    acc = wmma_f32(*(const v2f*)(hlr + k + half * 2),
                   *(const v2f*)(brow + Ee + k + half * 2), acc);
  float bias = bh[n];
  for (int j = 0; j < 8; ++j) {
    int row = m0 + j + 8 * half;
    int idx = row * Hh + n;
    float hhv = tanhf(acc[j] + bias);
    float zv = zb[idx];
    float hv = (1.0f - zv) * hl[idx] + zv * hhv;
    h_out[idx] = hv;
    hs_s[idx] = hv;
  }
}

// ------------------------------------------------------------- f32 -> bf16
__global__ __launch_bounds__(256) void k_cvt(const float* __restrict__ src,
                                             __bf16* __restrict__ dst, int n) {
  for (int i = blockIdx.x * 256 + threadIdx.x; i < n; i += gridDim.x * 256)
    dst[i] = (__bf16)src[i];
}

// --- out = hs_bf16 @ Wo^T + bo, sync-staged fallback (16-col tile, f32 Wo)
__global__ __launch_bounds__(256) void k_out_sync(
    const __bf16* __restrict__ hsb, const float* __restrict__ Wo,
    const float* __restrict__ bo, float* __restrict__ out) {
  __shared__ __bf16 Bs[16][Hh + 8];
  const int n0 = blockIdx.x * 16;                 // 2000 blocks over V
  const int wave = threadIdx.x >> 5, lane = threadIdx.x & 31;
  const int half = lane >> 4, l16 = lane & 15;
  for (int e = threadIdx.x; e < 16 * Hh; e += 256) {
    int row = e >> 10, col = e & (Hh - 1);
    Bs[row][col] = (__bf16)Wo[(size_t)(n0 + row) * Hh + col];
  }
  __syncthreads();
  const float bias = bo[n0 + l16];
  for (int mt = wave; mt < 64; mt += 8) {
    const int m0 = mt * 16;
    const __bf16* arow = hsb + (size_t)(m0 + l16) * Hh;
    v8f acc = {};
    for (int k = 0; k < Hh; k += 32) {
      v8bf alo = *(const v8bf*)(arow + k + half * 8);
      v8bf ahi = *(const v8bf*)(arow + k + 16 + half * 8);
      v8bf blo = *(const v8bf*)(&Bs[l16][k + half * 8]);
      v8bf bhi = *(const v8bf*)(&Bs[l16][k + 16 + half * 8]);
      acc = wmma_bf16(cat16(alo, ahi), cat16(blo, bhi), acc);
    }
    for (int j = 0; j < 8; ++j) {
      int row = m0 + j + 8 * half;
      out[(size_t)row * Vv + n0 + l16] = acc[j] + bias;
    }
  }
}

// --- out = hs_bf16 @ Wob^T + bo : 32-col tile async-staged via
//     GLOBAL_LOAD_ASYNC_TO_LDS_B128, 2x2 register blocking (4 wmma / k-step)
__global__ __launch_bounds__(256) void k_out_async(
    const __bf16* __restrict__ hsb, const __bf16* __restrict__ Wob,
    const float* __restrict__ bo, float* __restrict__ out) {
  __shared__ __bf16 Bs[32][Hh + 8];               // 66KB of 320KB WGP LDS
  const int n0 = blockIdx.x * 32;                 // 1000 blocks over V
  const int wave = threadIdx.x >> 5, lane = threadIdx.x & 31;
  const int half = lane >> 4, l16 = lane & 15;
  // async memory->LDS copy of the 32x1024 bf16 Wo tile (ASYNCcnt-tracked)
  for (int e = threadIdx.x; e < 32 * (Hh / 8); e += 256) {
    int row = e >> 7;                             // 128 16B chunks per row
    int col = (e & 127) * 8;
    unsigned lo = lds_offset(&Bs[row][col]);
    unsigned long long ga =
        (unsigned long long)(const void*)(Wob + (size_t)(n0 + row) * Hh + col);
    asm volatile("global_load_async_to_lds_b128 %0, %1, off"
                 :: "v"(lo), "v"(ga) : "memory");
  }
  asm volatile("s_wait_asynccnt 0x0" ::: "memory");
  __syncthreads();
  const float bias0 = bo[n0 + l16];
  const float bias1 = bo[n0 + 16 + l16];
  for (int mp = wave; mp < 32; mp += 8) {         // 8 waves sweep 32 M-pairs
    const int m0 = mp * 32;
    const __bf16* ar0 = hsb + (size_t)(m0 + l16) * Hh;
    const __bf16* ar1 = hsb + (size_t)(m0 + 16 + l16) * Hh;
    v8f a00 = {}, a01 = {}, a10 = {}, a11 = {};
    for (int k = 0; k < Hh; k += 32) {
      v16bf A0 = cat16(*(const v8bf*)(ar0 + k + half * 8),
                       *(const v8bf*)(ar0 + k + 16 + half * 8));
      v16bf A1 = cat16(*(const v8bf*)(ar1 + k + half * 8),
                       *(const v8bf*)(ar1 + k + 16 + half * 8));
      v16bf B0 = cat16(*(const v8bf*)(&Bs[l16][k + half * 8]),
                       *(const v8bf*)(&Bs[l16][k + 16 + half * 8]));
      v16bf B1 = cat16(*(const v8bf*)(&Bs[16 + l16][k + half * 8]),
                       *(const v8bf*)(&Bs[16 + l16][k + 16 + half * 8]));
      a00 = wmma_bf16(A0, B0, a00);
      a01 = wmma_bf16(A0, B1, a01);
      a10 = wmma_bf16(A1, B0, a10);
      a11 = wmma_bf16(A1, B1, a11);
    }
    for (int j = 0; j < 8; ++j) {
      int r0 = m0 + j + 8 * half, r1 = m0 + 16 + j + 8 * half;
      out[(size_t)r0 * Vv + n0 + l16]      = a00[j] + bias0;
      out[(size_t)r0 * Vv + n0 + 16 + l16] = a01[j] + bias1;
      out[(size_t)r1 * Vv + n0 + l16]      = a10[j] + bias0;
      out[(size_t)r1 * Vv + n0 + 16 + l16] = a11[j] + bias1;
    }
  }
}

extern "C" void kernel_launch(void* const* d_in, const int* in_sizes, int n_in,
                              void* d_out, int out_size, void* d_ws,
                              size_t ws_size, hipStream_t stream) {
  const int*   word = (const int*)d_in[0];
  const int*   seq  = (const int*)d_in[1];
  const float* emb  = (const float*)d_in[2];
  const float* Wl   = (const float*)d_in[3];
  const float* bl   = (const float*)d_in[4];
  const float* Wih  = (const float*)d_in[5];
  const float* Whh  = (const float*)d_in[6];
  const float* bih  = (const float*)d_in[7];
  const float* bhh  = (const float*)d_in[8];
  const float* Wz   = (const float*)d_in[9];
  const float* bz   = (const float*)d_in[10];
  const float* Wr   = (const float*)d_in[11];
  const float* br   = (const float*)d_in[12];
  const float* Wh   = (const float*)d_in[13];
  const float* bh   = (const float*)d_in[14];
  const float* Wo   = (const float*)d_in[15];
  const float* bo   = (const float*)d_in[16];
  float* outp = (float*)d_out;

  float* ws  = (float*)d_ws;
  float* we  = ws;                   // B*E      = 16384
  float* h   = we + Bb * Ee;         // B*H      = 32768
  float* c   = h + Bb * Hh;
  float* hl  = c + Bb * Hh;
  float* z   = hl + Bb * Hh;
  float* r   = z + Bb * Hh;          // B*E
  float* hs  = r + Bb * Ee;          // S*B*H    = 1048576
  __bf16* hsb = (__bf16*)(hs + Ss * Bb * Hh);                 // S*B*H bf16
  __bf16* Wob = (__bf16*)(hsb + Ss * Bb * Hh);                // V*H bf16
  const size_t need_async =
      (size_t)((char*)(Wob + (size_t)Vv * Hh) - (char*)d_ws);

  k_gather_we<<<64, 256, 0, stream>>>(emb, word, we);
  k_init<<<Hh / 16, 64, 0, stream>>>(we, Wl, bl, h, c);
  for (int s = 0; s < Ss; ++s) {
    k_lstm<<<Hh / 16, 128, 0, stream>>>(emb, seq, s, h, c, Wih, Whh, bih, bhh, hl);
    k_zr<<<96, 64, 0, stream>>>(we, hl, Wz, bz, Wr, br, z, r);
    k_hh<<<Hh / 16, 64, 0, stream>>>(we, hl, r, z, Wh, bh, h, hs + s * Bb * Hh);
  }
  k_cvt<<<1024, 256, 0, stream>>>(hs, hsb, Ss * Bb * Hh);

  if (ws_size >= need_async) {
    // pre-convert Wo to bf16 once: GEMM then reads Wo exactly once from HBM
    k_cvt<<<8192, 256, 0, stream>>>(Wo, Wob, Vv * Hh);
    k_out_async<<<Vv / 32, 256, 0, stream>>>(hsb, Wob, bo, outp);
  } else {
    k_out_sync<<<Vv / 16, 256, 0, stream>>>(hsb, Wo, bo, outp);
  }

  // hf, cf appended after logits
  hipMemcpyAsync(outp + (size_t)Ss * Bb * Vv, h, Bb * Hh * sizeof(float),
                 hipMemcpyDeviceToDevice, stream);
  hipMemcpyAsync(outp + (size_t)Ss * Bb * Vv + Bb * Hh, c,
                 Bb * Hh * sizeof(float), hipMemcpyDeviceToDevice, stream);
}
